// SelfAttention_32409823216366
// MI455X (gfx1250) — compile-verified, matrix operand order checked by reading, not run on hardware
//
#include <hip/hip_runtime.h>

typedef __attribute__((ext_vector_type(2)))  float    v2f;
typedef __attribute__((ext_vector_type(8)))  float    v8f;
typedef __attribute__((ext_vector_type(16))) _Float16 v16h;
typedef __attribute__((ext_vector_type(8)))  int      v8i;

union V16H { v8i i; v16h h; };

#define BATCH 8
#define NPIX  4096      // 64*64
#define CCH   64
#define DPROJ 8         // C/8

// ---------------------------------------------------------------------------
// Kernel 1: fused 1x1-conv projections.
//   f,g -> f32 row-major [B, N, 8]
//   h   -> f16, pre-swizzled into the WMMA B-fragment layout:
//          hswz[b][kc(32-key chunk)][cb(16-chan block)][lane][v(8 dwords)]
//          with k(v,half,lane) = 16*(v/4) + 8*(lane/16) + 2*(v%4) + half
// ---------------------------------------------------------------------------
__global__ __launch_bounds__(256) void sagan_proj_kernel(
    const float* __restrict__ x,  const float* __restrict__ kf,
    const float* __restrict__ kg, const float* __restrict__ kh,
    float* __restrict__ fbuf, float* __restrict__ gbuf,
    unsigned int* __restrict__ hswz)
{
    __shared__ float xs[64 * 64];    // 64 rows of x
    __shared__ float khs[64 * 64];   // kernel_h [in][out]
    __shared__ float kfg[64 * 16];   // [in][f0..7 | g0..7]

    const int t   = threadIdx.x;
    const int blk = blockIdx.x;          // 512 blocks
    const int b   = blk >> 6;            // batch
    const int bib = blk & 63;            // 64-row tile within batch
    const int rowbase = b * NPIX + bib * 64;

    {
        const float4* xg  = (const float4*)(x + rowbase * CCH);
        const float4* khg = (const float4*)kh;
        float4* xs4  = (float4*)xs;
        float4* khs4 = (float4*)khs;
#pragma unroll
        for (int i = 0; i < 4; ++i) {
            xs4[t + 256 * i]  = xg[t + 256 * i];
            khs4[t + 256 * i] = khg[t + 256 * i];
        }
    }
    if (t < 128) {
        const int in = t >> 1, j4 = (t & 1) * 4;
#pragma unroll
        for (int j = 0; j < 4; ++j) {
            kfg[in * 16 + j4 + j]     = kf[in * 8 + j4 + j];
            kfg[in * 16 + 8 + j4 + j] = kg[in * 8 + j4 + j];
        }
    }
    __syncthreads();

    // f and g: 64 rows x 16 outputs, 4 per thread
#pragma unroll
    for (int i = 0; i < 4; ++i) {
        const int idx = t + 256 * i;
        const int row = idx >> 4, col = idx & 15;
        float acc = 0.f;
#pragma unroll 8
        for (int c = 0; c < 64; ++c) acc += xs[row * 64 + c] * kfg[c * 16 + col];
        float* dst = (col < 8) ? fbuf : gbuf;
        dst[(rowbase + row) * DPROJ + (col & 7)] = acc;
    }

    // h, written directly in B-fragment order (16 f16 values per thread)
    {
        const int kc_local = t >> 7;        // 0..1  (32-key chunk in tile)
        const int cb       = (t >> 5) & 3;  // 0..3  (16-chan block)
        const int lane     = t & 31;
        unsigned int dw[8];
#pragma unroll
        for (int v = 0; v < 8; ++v) {
            const int kb = kc_local * 32 + ((v >> 2) << 4) + ((lane >> 4) << 3) + ((v & 3) << 1);
            const int ch = cb * 16 + (lane & 15);
            float a0 = 0.f, a1 = 0.f;
#pragma unroll 8
            for (int c = 0; c < 64; ++c) {
                const float w = khs[c * 64 + ch];
                a0 += xs[kb * 64 + c] * w;
                a1 += xs[(kb + 1) * 64 + c] * w;
            }
            const _Float16 h0 = (_Float16)a0, h1 = (_Float16)a1;
            const unsigned short u0 = __builtin_bit_cast(unsigned short, h0);
            const unsigned short u1 = __builtin_bit_cast(unsigned short, h1);
            dw[v] = (unsigned)u0 | ((unsigned)u1 << 16);
        }
        const int kc   = bib * 2 + kc_local;
        const int base = (((b * 128 + kc) * 4 + cb) * 32 + lane) * 8;
        uint4* dst = (uint4*)(hswz + base);
        uint4 s0; s0.x = dw[0]; s0.y = dw[1]; s0.z = dw[2]; s0.w = dw[3];
        uint4 s1; s1.x = dw[4]; s1.y = dw[5]; s1.z = dw[6]; s1.w = dw[7];
        dst[0] = s0;
        dst[1] = s1;
    }
}

// ---------------------------------------------------------------------------
// Kernel 2: fused two-pass flash attention + gamma residual.
//   grid = B * (N/128) blocks, 256 threads (8 wave32); wave = 16 queries.
//   Scores computed TRANSPOSED (s' = f @ g^T): keys in v-slots, queries in
//   lanes -> per-lane softmax stats, and P -> f16 A-fragment is a pure
//   register permutation (no LDS, no cross-lane traffic).
// ---------------------------------------------------------------------------
__global__ __launch_bounds__(256) void sagan_attn_kernel(
    const float* __restrict__ x,
    const float* __restrict__ fbuf, const float* __restrict__ gbuf,
    const unsigned int* __restrict__ hswz,
    const float* __restrict__ gamma_p, float* __restrict__ out)
{
    __shared__ float linv[8 * 16];          // per-wave 1/l exchange

    const int t    = threadIdx.x;
    const int wave = t >> 5;
    const int lane = t & 31;
    const int b    = blockIdx.x >> 5;       // 32 query-blocks per batch
    const int qblk = blockIdx.x & 31;
    const int q0   = qblk * 128 + wave * 16;
    const int qm   = lane & 15;             // lane%16: query col / key row
    const int hi   = lane >> 4;             // lane half

    // g as loop-invariant B-fragments: B[d][n] = g[q0+n][d]
    v2f gB0, gB1;
    {
        const int gb = (b * NPIX + q0 + qm) * DPROJ + 2 * hi;
        gB0.x = gbuf[gb + 0]; gB0.y = gbuf[gb + 1];
        gB1.x = gbuf[gb + 4]; gB1.y = gbuf[gb + 5];
    }

    const int fbase = (b * NPIX + qm) * DPROJ + 2 * hi;

    // ---------------- pass A: row max (per-lane, keys live in v-slots) -----
    float mloc = -__builtin_inff();
    for (int kc = 0; kc < 128; ++kc) {
#pragma unroll
        for (int tile = 0; tile < 2; ++tile) {
            const int fb = fbase + (kc * 32 + tile * 16) * DPROJ;
            v2f fA0, fA1;
            fA0.x = fbuf[fb + 0]; fA0.y = fbuf[fb + 1];
            fA1.x = fbuf[fb + 4]; fA1.y = fbuf[fb + 5];
            v8f s = (v8f){0.f,0.f,0.f,0.f,0.f,0.f,0.f,0.f};
            s = __builtin_amdgcn_wmma_f32_16x16x4_f32(false, fA0, false, gB0,
                                                      (short)0, s, false, false);
            s = __builtin_amdgcn_wmma_f32_16x16x4_f32(false, fA1, false, gB1,
                                                      (short)0, s, false, false);
#pragma unroll
            for (int v = 0; v < 8; ++v) mloc = fmaxf(mloc, s[v]);
        }
    }
    const float mfull = fmaxf(mloc, __shfl_xor(mloc, 16, 32));

    // ---------------- pass B: exp + accumulate (no rescaling needed) -------
    v8f acc[4];
#pragma unroll
    for (int j = 0; j < 4; ++j) acc[j] = (v8f){0.f,0.f,0.f,0.f,0.f,0.f,0.f,0.f};
    float lloc = 0.f;

    for (int kc = 0; kc < 128; ++kc) {
        float pt[2][8];
#pragma unroll
        for (int tile = 0; tile < 2; ++tile) {
            const int fb = fbase + (kc * 32 + tile * 16) * DPROJ;
            v2f fA0, fA1;
            fA0.x = fbuf[fb + 0]; fA0.y = fbuf[fb + 1];
            fA1.x = fbuf[fb + 4]; fA1.y = fbuf[fb + 5];
            v8f s = (v8f){0.f,0.f,0.f,0.f,0.f,0.f,0.f,0.f};
            s = __builtin_amdgcn_wmma_f32_16x16x4_f32(false, fA0, false, gB0,
                                                      (short)0, s, false, false);
            s = __builtin_amdgcn_wmma_f32_16x16x4_f32(false, fA1, false, gB1,
                                                      (short)0, s, false, false);
#pragma unroll
            for (int v = 0; v < 8; ++v) {
                const float p = __expf(s[v] - mfull);
                pt[tile][v] = p;
                lloc += p;
            }
        }

        // P' D-tiles -> f16 A-fragment: pure in-lane permutation + pack
        V16H pf;
#pragma unroll
        for (int v = 0; v < 8; ++v) {
            const int tile = v >> 2;
            const int r    = (v & 3) << 1;
            pf.i[v] = __builtin_bit_cast(int,
                __builtin_amdgcn_cvt_pkrtz(pt[tile][r], pt[tile][r + 1]));
        }

        // o += P @ h  (4 channel blocks, f16 WMMA, f32 accumulate)
        const int hb = ((b * 128 + kc) * 4) * 32 * 8;
#pragma unroll
        for (int j = 0; j < 4; ++j) {
            V16H hf;
            hf.i = *(const v8i*)(hswz + hb + (j * 32 + lane) * 8);
            acc[j] = __builtin_amdgcn_wmma_f32_16x16x32_f16(false, pf.h, false, hf.h,
                                                            (short)0, acc[j], false, false);
        }
    }

    // merge halves, move 1/l from lane-indexed to v-slot-indexed layout
    const float lfull = lloc + __shfl_xor(lloc, 16, 32);
    if (hi == 0) linv[wave * 16 + qm] = 1.0f / lfull;
    __syncthreads();

    // ---- normalize, gamma-residual, store ----
    const float gamma = gamma_p[0];
#pragma unroll
    for (int v = 0; v < 8; ++v) {
        const int   row = v + 8 * hi;       // query within the wave's 16
        const float rl  = linv[wave * 16 + row];
        const int   ob  = (b * NPIX + q0 + row) * CCH + qm;
#pragma unroll
        for (int j = 0; j < 4; ++j) {
            const float o = acc[j][v] * rl;
            out[ob + j * 16] = gamma * o + x[ob + j * 16];
        }
    }
}

// ---------------------------------------------------------------------------
extern "C" void kernel_launch(void* const* d_in, const int* in_sizes, int n_in,
                              void* d_out, int out_size, void* d_ws, size_t ws_size,
                              hipStream_t stream) {
    const float* x     = (const float*)d_in[0];
    const float* kf    = (const float*)d_in[1];
    const float* kg    = (const float*)d_in[2];
    const float* kh    = (const float*)d_in[3];
    const float* gamma = (const float*)d_in[4];
    float* out = (float*)d_out;

    // workspace: f (1MB f32) | g (1MB f32) | h swizzled f16 (4MB)
    float* fbuf = (float*)d_ws;
    float* gbuf = fbuf + BATCH * NPIX * DPROJ;
    unsigned int* hswz = (unsigned int*)(gbuf + BATCH * NPIX * DPROJ);

    sagan_proj_kernel<<<512, 256, 0, stream>>>(x, kf, kg, kh, fbuf, gbuf, hswz);
    sagan_attn_kernel<<<BATCH * (NPIX / 128), 256, 0, stream>>>(x, fbuf, gbuf, hswz,
                                                                gamma, out);
}
